// GRU_34454227649190
// MI455X (gfx1250) — matrix operand order repaired and checked by reading, not compiled, not run on hardware
//
#include <hip/hip_runtime.h>
#include <hip/hip_bf16.h>

typedef __attribute__((ext_vector_type(16))) _Float16 half16;
typedef __attribute__((ext_vector_type(8)))  _Float16 half8;
typedef __attribute__((ext_vector_type(8)))  float    float8;

#define T_LEN 512
#define BATCH 32
#define HID   512
#define G3    1536            // 3*HID
#define HSTR  (HID + 8)       // padded f16 h row stride (16B-aligned, banks staggered)

// ---------------------------------------------------------------------------
// WMMA wrapper: D(16x16,f32) = A(16x32,f16) * B(32x16,f16) + C
// ---------------------------------------------------------------------------
__device__ __forceinline__ float8 wmma_f16(half16 a, half16 b, float8 c) {
  return __builtin_amdgcn_wmma_f32_16x16x32_f16(
      /*neg_a=*/false, a, /*neg_b=*/false, b,
      /*c_mod=*/(short)0, c, /*reuse_a=*/false, /*reuse_b=*/false);
}

__device__ __forceinline__ float8 zero8() {
  float8 z = {0.f, 0.f, 0.f, 0.f, 0.f, 0.f, 0.f, 0.f};
  return z;
}

// ---------------------------------------------------------------------------
// Operand loaders (wave32). ISA 16-bit A 16x32 layout:
//   lanes 0-15 : row M = lane,    elems 0-7 -> K=k0..k0+7,   8-15 -> K=k0+16..k0+23
//   lanes 16-31: row M = lane-16, elems 0-7 -> K=k0+8..k0+15,8-15 -> K=k0+24..k0+31
// B 32x16: lane = column N; same K order as A (lanes 0-15: k0..k0+7, k0+16..k0+23; lanes 16-31: k0+8..k0+15, k0+24..k0+31).
// ---------------------------------------------------------------------------
__device__ __forceinline__ half16 load_A_g(const _Float16* __restrict__ p,
                                           int stride, int row0, int k0) {
  const int lane = threadIdx.x & 31;
  const int r  = row0 + (lane & 15);
  const int kb = k0 + ((lane >> 4) << 3);
  const half8* q0 = (const half8*)(p + (size_t)r * stride + kb);
  const half8* q1 = (const half8*)(p + (size_t)r * stride + kb + 16);
  half8 lo = *q0, hi = *q1;
  half16 v;
#pragma unroll
  for (int i = 0; i < 8; ++i) { v[i] = lo[i]; v[i + 8] = hi[i]; }
  return v;
}

__device__ __forceinline__ half16 load_B_g(const _Float16* __restrict__ p,
                                           int stride, int n0, int k0) {
  const int lane = threadIdx.x & 31;
  const int r  = n0 + (lane & 15);
  const int kb = k0 + ((lane >> 4) << 3);
  const half8* q0 = (const half8*)(p + (size_t)r * stride + kb);
  const half8* q1 = (const half8*)(p + (size_t)r * stride + kb + 16);
  half8 lo = *q0, hi = *q1;
  half16 v;
#pragma unroll
  for (int i = 0; i < 8; ++i) { v[i] = lo[i]; v[i + 8] = hi[i]; }
  return v;
}

// A operand from the f16 hidden-state mirror in LDS (padded stride HSTR).
__device__ __forceinline__ half16 load_A_ldsH(const _Float16* p, int row0, int k0) {
  const int lane = threadIdx.x & 31;
  const int r  = row0 + (lane & 15);
  const int kb = k0 + ((lane >> 4) << 3);
  const half8* q0 = (const half8*)(p + r * HSTR + kb);
  const half8* q1 = (const half8*)(p + r * HSTR + kb + 16);
  half8 lo = *q0, hi = *q1;
  half16 v;
#pragma unroll
  for (int i = 0; i < 8; ++i) { v[i] = lo[i]; v[i + 8] = hi[i]; }
  return v;
}

// ---------------------------------------------------------------------------
// Conversion kernels
// ---------------------------------------------------------------------------
__global__ __launch_bounds__(256) void cvt_f32_to_f16(const float* __restrict__ in,
                                                      _Float16* __restrict__ out, int n) {
  int i = blockIdx.x * 256 + threadIdx.x;
  if (i < n) out[i] = (_Float16)in[i];
}

// x[B][T][F] -> xT[T][B][F] as f16 (F = 256)
__global__ __launch_bounds__(256) void cvt_x_kernel(const float* __restrict__ x,
                                                    _Float16* __restrict__ xT) {
  int i = blockIdx.x * 256 + threadIdx.x;            // 32*512*256 = 4194304
  if (i >= BATCH * T_LEN * 256) return;
  int f = i & 255;
  int t = (i >> 8) & 511;
  int b = i >> 17;
  xT[(size_t)(t * BATCH + b) * 256 + f] = (_Float16)x[i];
}

// ---------------------------------------------------------------------------
// gx = act[M,K] * W[dir][3H,K]^T + b_ih[dir]   (M = T*B, row = t*B+b)
// One 16x16 output tile per wave; 8 waves per block; blockIdx.y = direction.
// ---------------------------------------------------------------------------
__global__ __launch_bounds__(256) void gx_gemm_kernel(
    const _Float16* __restrict__ act, const _Float16* __restrict__ wih,
    const float* __restrict__ b_ih, float* __restrict__ gx, int M, int K) {
  const int dir  = blockIdx.y;
  const int wave = threadIdx.x >> 5;
  const int lane = threadIdx.x & 31;
  const int tile = blockIdx.x * 8 + wave;
  const int tn = tile % (G3 / 16);
  const int tm = tile / (G3 / 16);
  if (tm >= (M >> 4)) return;   // wave-uniform

  const _Float16* wp = wih + (size_t)dir * G3 * K;
  float8 acc = zero8();
  for (int k = 0; k < K; k += 32) {
    half16 a = load_A_g(act, K, tm << 4, k);
    half16 b = load_B_g(wp,  K, tn << 4, k);
    acc = wmma_f16(a, b, acc);
  }
  const int n = (tn << 4) + (lane & 15);
  const float bias = b_ih[dir * G3 + n];
  float* o = gx + (size_t)dir * M * G3;
#pragma unroll
  for (int e = 0; e < 8; ++e) {
    int m = (tm << 4) + e + ((lane >> 4) << 3);
    o[(size_t)m * G3 + n] = acc[e] + bias;
  }
}

// ---------------------------------------------------------------------------
// Persistent recurrent scan. grid = {2} (dir), block = 1024 (32 waves).
// Wave j owns hidden columns [16j, 16j+16); per step it computes the r/z/n
// 16x16 tiles for both batch halves (6 WMMA accumulators), applies the gates
// entirely in registers, and updates h. h is kept twice in LDS: f32 master
// (for the exact z*h blend) and a padded f16 mirror (WMMA A operand, loaded
// with ds_load_b128, converted only once per element per timestep).
// ---------------------------------------------------------------------------
__global__ __launch_bounds__(1024) void gru_scan_kernel(
    const float* __restrict__ gx,      // [2][T*B][3H]
    const _Float16* __restrict__ w_hh, // [2][3H][H] f16
    const float* __restrict__ b_hh,    // [2][3H]
    _Float16* __restrict__ ys_h,       // [T][B][2H] f16 (layer0) or null
    float* __restrict__ ys_f,          // [B][T][2H] f32 (layer1) or null
    float* __restrict__ hn) {          // [2][B][H]
  __shared__ float    hF[BATCH * HID];    // 64 KB, exact f32 state
  __shared__ _Float16 hH[BATCH * HSTR];   // ~32.5 KB, f16 mirror (padded rows)

  const int dir  = blockIdx.x;
  const int wave = threadIdx.x >> 5;
  const int lane = threadIdx.x & 31;

  for (int i = threadIdx.x; i < BATCH * HID; i += 1024) hF[i] = 0.f;
  for (int i = threadIdx.x; i < BATCH * HSTR; i += 1024) hH[i] = (_Float16)0.f;
  __syncthreads();

  const int n0   = wave << 4;
  const int ncol = n0 + (lane & 15);            // hidden unit for epilogue lanes
  const _Float16* wp  = w_hh + (size_t)dir * G3 * HID;
  const float*    gxd = gx   + (size_t)dir * T_LEN * BATCH * G3;
  const float br = b_hh[dir * G3 + ncol];
  const float bz = b_hh[dir * G3 + HID + ncol];
  const float bn = b_hh[dir * G3 + 2 * HID + ncol];

  for (int s = 0; s < T_LEN; ++s) {
    const int t = dir ? (T_LEN - 1 - s) : s;

    // Prefetch next timestep's gx slab (spread across the workgroup).
    {
      const int tn = dir ? (t > 0 ? t - 1 : 0) : (t < T_LEN - 1 ? t + 1 : t);
      __builtin_prefetch(gxd + (size_t)tn * BATCH * G3 + (size_t)threadIdx.x * 48, 0, 0);
    }

    float8 ar0 = zero8(), ar1 = zero8();
    float8 az0 = zero8(), az1 = zero8();
    float8 an0 = zero8(), an1 = zero8();

    for (int k = 0; k < HID; k += 32) {
      half16 a0  = load_A_ldsH(hH, 0,  k);
      half16 a1  = load_A_ldsH(hH, 16, k);
      half16 brw = load_B_g(wp, HID, n0,           k);
      half16 bzw = load_B_g(wp, HID, HID + n0,     k);
      half16 bnw = load_B_g(wp, HID, 2 * HID + n0, k);
      ar0 = wmma_f16(a0, brw, ar0);  ar1 = wmma_f16(a1, brw, ar1);
      az0 = wmma_f16(a0, bzw, az0);  az1 = wmma_f16(a1, bzw, az1);
      an0 = wmma_f16(a0, bnw, an0);  an1 = wmma_f16(a1, bnw, an1);
    }

    const float* gxt = gxd + (size_t)t * BATCH * G3;
    float hnew[16];
#pragma unroll
    for (int h2 = 0; h2 < 2; ++h2) {
#pragma unroll
      for (int e = 0; e < 8; ++e) {
        const int b = (h2 << 4) + e + ((lane >> 4) << 3);
        const float R = (h2 ? ar1[e] : ar0[e]) + br;
        const float Z = (h2 ? az1[e] : az0[e]) + bz;
        const float N = (h2 ? an1[e] : an0[e]) + bn;
        const float xr = gxt[(size_t)b * G3 + ncol];
        const float xz = gxt[(size_t)b * G3 + HID + ncol];
        const float xn = gxt[(size_t)b * G3 + 2 * HID + ncol];
        const float rg = 1.f / (1.f + __expf(-(xr + R)));
        const float zg = 1.f / (1.f + __expf(-(xz + Z)));
        const float ng = tanhf(xn + rg * N);
        const float ho = hF[b * HID + ncol];
        hnew[(h2 << 3) + e] = (1.f - zg) * ng + zg * ho;
      }
    }
    __syncthreads();   // all reads of hF/hH for this step complete
#pragma unroll
    for (int h2 = 0; h2 < 2; ++h2) {
#pragma unroll
      for (int e = 0; e < 8; ++e) {
        const int b = (h2 << 4) + e + ((lane >> 4) << 3);
        const float v = hnew[(h2 << 3) + e];
        hF[b * HID + ncol]  = v;
        hH[b * HSTR + ncol] = (_Float16)v;
        if (ys_h) ys_h[((size_t)t * BATCH + b) * (2 * HID) + (size_t)dir * HID + ncol] = (_Float16)v;
        else      ys_f[((size_t)b * T_LEN + t) * (2 * HID) + (size_t)dir * HID + ncol] = v;
      }
    }
    __syncthreads();   // h update visible before next step
  }

  for (int i = threadIdx.x; i < BATCH * HID; i += 1024)
    hn[(size_t)dir * BATCH * HID + i] = hF[i];
}

// ---------------------------------------------------------------------------
// Host-side launcher
// ---------------------------------------------------------------------------
extern "C" void kernel_launch(void* const* d_in, const int* in_sizes, int n_in,
                              void* d_out, int out_size, void* d_ws, size_t ws_size,
                              hipStream_t stream) {
  const float* x     = (const float*)d_in[0];
  const float* w_ih0 = (const float*)d_in[1];
  const float* w_hh0 = (const float*)d_in[2];
  const float* b_ih0 = (const float*)d_in[3];
  const float* b_hh0 = (const float*)d_in[4];
  const float* w_ih1 = (const float*)d_in[5];
  const float* w_hh1 = (const float*)d_in[6];
  const float* b_ih1 = (const float*)d_in[7];
  const float* b_hh1 = (const float*)d_in[8];

  float* out = (float*)d_out;                               // [B][T][2H]
  float* hn  = out + (size_t)BATCH * T_LEN * 2 * HID;       // [4][B][H]

  // Workspace carve-up (all sizes 256B-aligned by construction).
  char* w = (char*)d_ws;
  _Float16* xT_h   = (_Float16*)w;  w += (size_t)T_LEN * BATCH * 256 * 2;   //  8 MB
  _Float16* wih0_h = (_Float16*)w;  w += (size_t)2 * G3 * 256  * 2;         //  1.5 MB
  _Float16* whh0_h = (_Float16*)w;  w += (size_t)2 * G3 * HID  * 2;         //  3 MB
  _Float16* wih1_h = (_Float16*)w;  w += (size_t)2 * G3 * 1024 * 2;         //  6 MB
  _Float16* whh1_h = (_Float16*)w;  w += (size_t)2 * G3 * HID  * 2;         //  3 MB
  _Float16* out0_h = (_Float16*)w;  w += (size_t)T_LEN * BATCH * 1024 * 2;  // 32 MB
  float*    gx0    = (float*)w;     w += (size_t)2 * T_LEN * BATCH * G3 * 4;// 201 MB
  float*    gx1    = (float*)w;     w += (size_t)2 * T_LEN * BATCH * G3 * 4;// 201 MB

  const int M = T_LEN * BATCH;   // 16384

  // 1) f32 -> f16 conversions
  cvt_x_kernel<<<(BATCH * T_LEN * 256 + 255) / 256, 256, 0, stream>>>(x, xT_h);
  cvt_f32_to_f16<<<(2 * G3 * 256  + 255) / 256, 256, 0, stream>>>(w_ih0, wih0_h, 2 * G3 * 256);
  cvt_f32_to_f16<<<(2 * G3 * HID  + 255) / 256, 256, 0, stream>>>(w_hh0, whh0_h, 2 * G3 * HID);
  cvt_f32_to_f16<<<(2 * G3 * 1024 + 255) / 256, 256, 0, stream>>>(w_ih1, wih1_h, 2 * G3 * 1024);
  cvt_f32_to_f16<<<(2 * G3 * HID  + 255) / 256, 256, 0, stream>>>(w_hh1, whh1_h, 2 * G3 * HID);

  const int tiles = (M / 16) * (G3 / 16);   // 98304, divisible by 8
  dim3 gxGrid(tiles / 8, 2);

  // 2) Layer 0: input projection + scan
  gx_gemm_kernel<<<gxGrid, 256, 0, stream>>>(xT_h, wih0_h, b_ih0, gx0, M, 256);
  gru_scan_kernel<<<2, 1024, 0, stream>>>(gx0, whh0_h, b_hh0, out0_h, nullptr, hn);

  // 3) Layer 1: input projection + scan
  gx_gemm_kernel<<<gxGrid, 256, 0, stream>>>(out0_h, wih1_h, b_ih1, gx1, M, 1024);
  gru_scan_kernel<<<2, 1024, 0, stream>>>(gx1, whh1_h, b_hh1, nullptr, out,
                                          hn + (size_t)2 * BATCH * HID);
}
